// Decoder_51247549776174
// MI455X (gfx1250) — matrix operand order repaired and checked by reading, not compiled, not run on hardware
//
#include <hip/hip_runtime.h>
#include <math.h>

// ---- problem constants (from reference) ----
constexpr int Dm  = 256;     // D
constexpr int Tm  = 1024;    // T
constexpr int Bm  = 64;      // B
constexpr int Lm  = 65;      // L
constexpr int Vm  = 4235;    // V
constexpr int Sm  = 64;      // L-1 scan steps
constexpr int G4  = 4 * Dm;  // 1024 gate width
constexpr int K2  = 2 * Dm;  // 512
constexpr int K3  = 3 * Dm;  // 768 classifier K
constexpr int CHR = 16;      // eout rows per TDM chunk in ctx loop

// ---- CDNA5 WMMA / descriptor types ----
typedef __attribute__((ext_vector_type(16))) __bf16 v16bf;
typedef __attribute__((ext_vector_type(8)))  float  v8f;
typedef __attribute__((ext_vector_type(4)))  unsigned int v4u;
typedef __attribute__((ext_vector_type(8)))  unsigned int v8u;

union BfFrag { unsigned int u[8]; v16bf v; };

__device__ __forceinline__ unsigned int pack_bf16x2(float a, float b) {
    unsigned int ua = __float_as_uint(a);
    unsigned int ub = __float_as_uint(b);
    ua = (ua + 0x7FFFu + ((ua >> 16) & 1u)) >> 16;   // RNE fp32 -> bf16
    ub = (ub + 0x7FFFu + ((ub >> 16) & 1u)) >> 16;
    return (ua & 0xFFFFu) | (ub << 16);
}
__device__ __forceinline__ unsigned short f2bf(float a) {
    unsigned int ua = __float_as_uint(a);
    return (unsigned short)((ua + 0x7FFFu + ((ua >> 16) & 1u)) >> 16);
}

// A-frag (16x32 bf16), bf16 row-major source; rowp = &src[row*ld], row = lane&15.
// lanes 0-15: K = kb+{0..7,16..23}; lanes 16-31: K = kb+{8..15,24..31}.
__device__ __forceinline__ v16bf load_a_bf(const unsigned short* rowp, int kb, int lane) {
    int g = (lane >> 4) * 8;
    const uint4* p0 = reinterpret_cast<const uint4*>(rowp + kb + g);
    const uint4* p1 = reinterpret_cast<const uint4*>(rowp + kb + 16 + g);
    uint4 a = p0[0], b = p1[0];
    BfFrag f;
    f.u[0] = a.x; f.u[1] = a.y; f.u[2] = a.z; f.u[3] = a.w;
    f.u[4] = b.x; f.u[5] = b.y; f.u[6] = b.z; f.u[7] = b.w;
    return f.v;
}
// B-frag (32x16 bf16) for D = A x B with B[k][n] = W[n][k], W bf16 row-major (N,K).
__device__ __forceinline__ v16bf load_b_bf(const unsigned short* W, int ldk, int n, int kb,
                                           int lane, bool valid) {
    BfFrag f;
    if (valid) {
        const uint4* p0 = reinterpret_cast<const uint4*>(W + (size_t)n * ldk + kb + ((lane >> 4) << 3));
        const uint4* p1 = reinterpret_cast<const uint4*>(W + (size_t)n * ldk + kb + 16 + ((lane >> 4) << 3));
        uint4 a = p0[0], b = p1[0];
        f.u[0] = a.x; f.u[1] = a.y; f.u[2] = a.z; f.u[3] = a.w;
        f.u[4] = b.x; f.u[5] = b.y; f.u[6] = b.z; f.u[7] = b.w;
    } else {
        #pragma unroll
        for (int j = 0; j < 8; ++j) f.u[j] = 0u;
    }
    return f.v;
}

__device__ __forceinline__ float sigf(float x) { return 1.0f / (1.0f + expf(-x)); }

// ---- TDM: 1-D tile of nelem fp32 from global -> LDS (2-group D#, wave-uniform args) ----
__device__ __forceinline__ void tdm_load_f32(const float* gsrc, unsigned lds_off, int nelem) {
    unsigned long long ga = (unsigned long long)(size_t)gsrc;
    v4u g0;
    g0[0] = 1u;                                               // count=1, user D#
    g0[1] = lds_off;                                          // lds_addr (bytes)
    g0[2] = (unsigned)ga;                                     // global_addr[31:0]
    g0[3] = (unsigned)((ga >> 32) & 0x01FFFFFFu) | (2u << 30);// global_addr[56:32] | type=2
    v8u g1;
    unsigned ne = (unsigned)nelem;
    g1[0] = (2u << 16);                                       // data_size=4B, no multicast
    g1[1] = (ne & 0xFFFFu) << 16;                             // tensor_dim0[15:0]
    g1[2] = ((ne >> 16) & 0xFFFFu) | (1u << 16);              // tensor_dim0[31:16] | tensor_dim1=1
    g1[3] = (ne & 0xFFFFu) << 16;                             // tile_dim0 = nelem
    g1[4] = 1u;                                               // tile_dim1=1, tile_dim2=0
    g1[5] = ne;                                               // tensor_dim0_stride[31:0]
    g1[6] = 0u;
    g1[7] = 0u;
    asm volatile("tensor_load_to_lds %0, %1" :: "s"(g0), "s"(g1) : "memory");
}

// ================= prep kernels =================
__global__ void k_transpose(const float* __restrict__ in, float* __restrict__ out, int R, int C) {
    int idx = blockIdx.x * blockDim.x + threadIdx.x;
    if (idx < R * C) {
        int r = idx / C, c = idx % C;
        out[(size_t)c * R + r] = in[idx];   // out[k][j] = in[j][k]
    }
}
// vectorized fp32 -> bf16 (n must be a multiple of 4; all users are)
__global__ void k_f32_to_bf16(const float* __restrict__ in, unsigned int* __restrict__ out, int n4) {
    int idx = blockIdx.x * blockDim.x + threadIdx.x;
    if (idx < n4) {
        float4 v = reinterpret_cast<const float4*>(in)[idx];
        uint2 o;
        o.x = pack_bf16x2(v.x, v.y);
        o.y = pack_bf16x2(v.z, v.w);
        reinterpret_cast<uint2*>(out)[idx] = o;
    }
}

// ================= kernel 1: att_h = eout @ wW.T + wb  (M=65536,N=256,K=256, bf16 WMMA) ===========
// block = 256 threads (8 waves): 4 M-waves x 2 N-waves; WG tile = 64 rows x 256 cols.
__global__ __launch_bounds__(256) void k_att_h(const unsigned short* __restrict__ eout_bf,
                                               const unsigned short* __restrict__ wW_bf,
                                               const float* __restrict__ wb,
                                               float* __restrict__ att_h) {
    const int lane = threadIdx.x & 31;
    const int wave = threadIdx.x >> 5;
    const int mw = wave & 3, nw = wave >> 2;
    const int mbase = blockIdx.x * 64 + mw * 16;
    const unsigned short* ap = eout_bf + (size_t)(mbase + (lane & 15)) * Dm;

    v8f acc[8];
    v8f zero = {};
    #pragma unroll
    for (int t = 0; t < 8; ++t) acc[t] = zero;

    for (int kb = 0; kb < Dm; kb += 32) {
        v16bf a = load_a_bf(ap, kb, lane);
        #pragma unroll
        for (int t = 0; t < 8; ++t) {
            int n = nw * 128 + t * 16 + (lane & 15);
            v16bf b = load_b_bf(wW_bf, Dm, n, kb, lane, true);
            acc[t] = __builtin_amdgcn_wmma_f32_16x16x32_bf16(false, a, false, b,
                                                             (short)0, acc[t], false, false);
        }
    }
    #pragma unroll
    for (int t = 0; t < 8; ++t) {
        int n = nw * 128 + t * 16 + (lane & 15);
        float bias = wb[n];
        #pragma unroll
        for (int j = 0; j < 8; ++j) {
            int row = mbase + (lane >> 4) * 8 + j;
            att_h[(size_t)row * Dm + n] = acc[t][j] + bias;
        }
    }
}

// ================= kernel 2: attention LSTM scan (one WG per batch row) =================
__global__ __launch_bounds__(256) void k_att_scan(
    const float* __restrict__ eout, const float* __restrict__ x_mask,
    const int* __restrict__ y, const float* __restrict__ y_mask,
    const float* __restrict__ emb,
    const float* __restrict__ WihT,   // (512,1024)  WihT[k][j] = att_Wih[j][k]
    const float* __restrict__ WhhT,   // (256,1024)
    const float* __restrict__ att_b,  // (1024)
    const float* __restrict__ vWT,    // (256,256)
    const float* __restrict__ vb,     // (256)
    const float* __restrict__ w_att_v,// (256)
    const float* __restrict__ att_h,  // (B,T,256)
    float* __restrict__ att_fea)      // (B,64,512)
{
    __shared__ float s_ebuf[2][CHR * Dm];        // TDM double buffer (2 x 16 KB)
    __shared__ float s_h[Dm], s_c[Dm], s_ctx[Dm], s_e[Dm], s_state[Dm], s_wv[Dm];
    __shared__ float s_g[G4];
    __shared__ float s_sc[Tm];
    __shared__ float s_red[256];

    const int b = blockIdx.x;
    const int tid = threadIdx.x;
    s_h[tid] = 0.f; s_c[tid] = 0.f; s_ctx[tid] = 0.f; s_wv[tid] = w_att_v[tid];
    __syncthreads();

    const float* atth_b = att_h + (size_t)b * Tm * Dm;
    const float* eout_b = eout + (size_t)b * Tm * Dm;
    const unsigned lds0 = (unsigned)(size_t)&s_ebuf[0][0];   // LDS aperture: low 32 bits = offset
    const unsigned lds1 = (unsigned)(size_t)&s_ebuf[1][0];

    for (int t = 0; t < Sm; ++t) {
        // ---- embedding lookup: e_t = emb[y[b,t]] ----
        int tok = y[b * Lm + t];
        s_e[tid] = emb[(size_t)tok * Dm + tid];
        __syncthreads();

        // ---- gates g = [e,ctx]@Wih.T + h@Whh.T + b ; thread owns j = tid*4..+3 ----
        float4 acc = *reinterpret_cast<const float4*>(att_b + tid * 4);
        #pragma unroll 4
        for (int k = 0; k < K2; ++k) {
            float xk = (k < Dm) ? s_e[k] : s_ctx[k - Dm];
            float4 w = *reinterpret_cast<const float4*>(WihT + (size_t)k * G4 + tid * 4);
            acc.x = fmaf(xk, w.x, acc.x); acc.y = fmaf(xk, w.y, acc.y);
            acc.z = fmaf(xk, w.z, acc.z); acc.w = fmaf(xk, w.w, acc.w);
        }
        #pragma unroll 4
        for (int k = 0; k < Dm; ++k) {
            float hk = s_h[k];
            float4 w = *reinterpret_cast<const float4*>(WhhT + (size_t)k * G4 + tid * 4);
            acc.x = fmaf(hk, w.x, acc.x); acc.y = fmaf(hk, w.y, acc.y);
            acc.z = fmaf(hk, w.z, acc.z); acc.w = fmaf(hk, w.w, acc.w);
        }
        reinterpret_cast<float4*>(s_g)[tid] = acc;
        __syncthreads();

        // ---- LSTM elementwise (g = [i,f,g,o]) ----
        {
            float gi = s_g[tid], gf = s_g[Dm + tid], gg = s_g[2 * Dm + tid], go = s_g[3 * Dm + tid];
            float cc = sigf(gf) * s_c[tid] + sigf(gi) * tanhf(gg);
            float hh = sigf(go) * tanhf(cc);
            s_c[tid] = cc; s_h[tid] = hh;
        }
        __syncthreads();

        // ---- state = h @ vW.T + vb ----
        {
            float a = vb[tid];
            #pragma unroll 4
            for (int k = 0; k < Dm; ++k) a = fmaf(s_h[k], vWT[(size_t)k * Dm + tid], a);
            s_state[tid] = a;
        }
        __syncthreads();

        // ---- scores over T: sum_d wv[d]*tanh(state[d]+att_h[b,tt,d]) + mbias ----
        #pragma unroll
        for (int q = 0; q < 4; ++q) {
            int tt = tid + q * 256;
            const float4* ph = reinterpret_cast<const float4*>(atth_b + (size_t)tt * Dm);
            __builtin_prefetch(atth_b + (size_t)(tt + 1) * Dm, 0, 0);
            float a = 0.f;
            #pragma unroll 4
            for (int d4 = 0; d4 < Dm / 4; ++d4) {
                float4 v = ph[d4];
                int d = d4 * 4;
                a = fmaf(s_wv[d + 0], tanhf(s_state[d + 0] + v.x), a);
                a = fmaf(s_wv[d + 1], tanhf(s_state[d + 1] + v.y), a);
                a = fmaf(s_wv[d + 2], tanhf(s_state[d + 2] + v.z), a);
                a = fmaf(s_wv[d + 3], tanhf(s_state[d + 3] + v.w), a);
            }
            a += (x_mask[(size_t)b * Tm + tt] - 1.0f) * 1e30f;
            s_sc[tt] = a;
        }
        __syncthreads();

        // ---- softmax over T ----
        float m4 = fmaxf(fmaxf(s_sc[tid], s_sc[tid + 256]), fmaxf(s_sc[tid + 512], s_sc[tid + 768]));
        s_red[tid] = m4; __syncthreads();
        for (int off = 128; off > 0; off >>= 1) {
            if (tid < off) s_red[tid] = fmaxf(s_red[tid], s_red[tid + off]);
            __syncthreads();
        }
        float mx = s_red[0]; __syncthreads();
        float lsum = 0.f;
        #pragma unroll
        for (int q = 0; q < 4; ++q) {
            int tt = tid + q * 256;
            float e = expf(s_sc[tt] - mx);
            s_sc[tt] = e; lsum += e;
        }
        s_red[tid] = lsum; __syncthreads();
        for (int off = 128; off > 0; off >>= 1) {
            if (tid < off) s_red[tid] += s_red[tid + off];
            __syncthreads();
        }
        float inv = 1.0f / s_red[0];
        __syncthreads();

        // ---- ctx = softmax(scores) @ eout[b] via TDM double-buffered LDS chunks ----
        {
            constexpr int NCH = Tm / CHR;                 // 64 chunks of 16 rows (16 KB each)
            if (threadIdx.x < 32) {                       // wave 0 drives the DMA
                tdm_load_f32(eout_b, lds0, CHR * Dm);
            }
            float a = 0.f;
            for (int c = 0; c < NCH; ++c) {
                if (threadIdx.x < 32) {
                    if (c + 1 < NCH) {
                        tdm_load_f32(eout_b + (size_t)(c + 1) * CHR * Dm,
                                     ((c + 1) & 1) ? lds1 : lds0, CHR * Dm);
                        __builtin_amdgcn_s_wait_tensorcnt(1);  // chunk c landed
                    } else {
                        __builtin_amdgcn_s_wait_tensorcnt(0);
                    }
                }
                __syncthreads();                           // publish chunk c to all waves
                const float* eb = s_ebuf[c & 1];
                const int sbase = c * CHR;
                #pragma unroll
                for (int r = 0; r < CHR; ++r)
                    a = fmaf(s_sc[sbase + r], eb[r * Dm + tid], a);
                __syncthreads();                           // done reading before re-fill
            }
            s_ctx[tid] = a * inv;
        }
        __syncthreads();

        // ---- emit att_fea = [h*ym, ctx*ym] ----
        float ym = y_mask[b * Lm + t + 1];
        float* fout = att_fea + (size_t)(b * Sm + t) * K2;
        fout[tid] = s_h[tid] * ym;
        fout[Dm + tid] = s_ctx[tid] * ym;
        __syncthreads();
    }
}

// ================= kernel 3: decoder LSTM scan -> dout (bf16) =================
__global__ __launch_bounds__(256) void k_dec_scan(
    const float* __restrict__ att_fea, const float* __restrict__ y_mask,
    const float* __restrict__ WihT,   // (512,1024) dec
    const float* __restrict__ WhhT,   // (256,1024) dec
    const float* __restrict__ dec_b,  // (1024)
    unsigned short* __restrict__ dout_bf) // (B*64, 768) bf16
{
    __shared__ float s_x[K2], s_h[Dm], s_c[Dm], s_g[G4];
    const int b = blockIdx.x;
    const int tid = threadIdx.x;
    s_h[tid] = 0.f; s_c[tid] = 0.f;
    __syncthreads();

    for (int t = 0; t < Sm; ++t) {
        const float* xin = att_fea + (size_t)(b * Sm + t) * K2;
        s_x[tid] = xin[tid];
        s_x[Dm + tid] = xin[Dm + tid];
        __syncthreads();

        float4 acc = *reinterpret_cast<const float4*>(dec_b + tid * 4);
        #pragma unroll 4
        for (int k = 0; k < K2; ++k) {
            float xk = s_x[k];
            float4 w = *reinterpret_cast<const float4*>(WihT + (size_t)k * G4 + tid * 4);
            acc.x = fmaf(xk, w.x, acc.x); acc.y = fmaf(xk, w.y, acc.y);
            acc.z = fmaf(xk, w.z, acc.z); acc.w = fmaf(xk, w.w, acc.w);
        }
        #pragma unroll 4
        for (int k = 0; k < Dm; ++k) {
            float hk = s_h[k];
            float4 w = *reinterpret_cast<const float4*>(WhhT + (size_t)k * G4 + tid * 4);
            acc.x = fmaf(hk, w.x, acc.x); acc.y = fmaf(hk, w.y, acc.y);
            acc.z = fmaf(hk, w.z, acc.z); acc.w = fmaf(hk, w.w, acc.w);
        }
        reinterpret_cast<float4*>(s_g)[tid] = acc;
        __syncthreads();

        float gi = s_g[tid], gf = s_g[Dm + tid], gg = s_g[2 * Dm + tid], go = s_g[3 * Dm + tid];
        float cc = sigf(gf) * s_c[tid] + sigf(gi) * tanhf(gg);
        float hh = sigf(go) * tanhf(cc);
        s_c[tid] = cc; s_h[tid] = hh;

        // dout = [att_fea*ym, dh*ym]  (att_fea already carries one ym; ref applies again)
        float ym = y_mask[b * Lm + t + 1];
        unsigned short* po = dout_bf + (size_t)(b * Sm + t) * K3;
        po[tid]          = f2bf(s_x[tid] * ym);
        po[Dm + tid]     = f2bf(s_x[Dm + tid] * ym);
        po[2 * Dm + tid] = f2bf(hh * ym);
        __syncthreads();
    }
}

// ================= kernel 4: logit = dout @ cls_W.T + cls_b  (M=4096,N=4235,K=768) ===============
__global__ __launch_bounds__(256) void k_cls(const unsigned short* __restrict__ dout_bf,
                                             const unsigned short* __restrict__ clsW_bf,
                                             const float* __restrict__ cls_b,
                                             float* __restrict__ logit) {
    const int lane = threadIdx.x & 31;
    const int wave = threadIdx.x >> 5;
    const int mw = wave & 3, nw = wave >> 2;
    const int mbase = blockIdx.x * 64 + mw * 16;
    const int nb0 = blockIdx.y * 256 + nw * 128;
    const unsigned short* ap = dout_bf + (size_t)(mbase + (lane & 15)) * K3;

    v8f acc[8];
    v8f zero = {};
    #pragma unroll
    for (int t = 0; t < 8; ++t) acc[t] = zero;

    for (int kb = 0; kb < K3; kb += 32) {
        v16bf a = load_a_bf(ap, kb, lane);
        #pragma unroll
        for (int t = 0; t < 8; ++t) {
            int n = nb0 + t * 16 + (lane & 15);
            v16bf b = load_b_bf(clsW_bf, K3, n, kb, lane, n < Vm);
            acc[t] = __builtin_amdgcn_wmma_f32_16x16x32_bf16(false, a, false, b,
                                                             (short)0, acc[t], false, false);
        }
    }
    #pragma unroll
    for (int t = 0; t < 8; ++t) {
        int n = nb0 + t * 16 + (lane & 15);
        if (n < Vm) {
            float bias = cls_b[n];
            #pragma unroll
            for (int j = 0; j < 8; ++j) {
                int row = mbase + (lane >> 4) * 8 + j;
                logit[(size_t)row * Vm + n] = acc[t][j] + bias;
            }
        }
    }
}

// ================= host launcher =================
extern "C" void kernel_launch(void* const* d_in, const int* in_sizes, int n_in,
                              void* d_out, int out_size, void* d_ws, size_t ws_size,
                              hipStream_t stream) {
    const float* eout    = (const float*)d_in[0];
    const float* x_mask  = (const float*)d_in[1];
    const int*   y       = (const int*)d_in[2];
    const float* y_mask  = (const float*)d_in[3];
    const float* emb     = (const float*)d_in[4];
    const float* att_Wih = (const float*)d_in[5];
    const float* att_Whh = (const float*)d_in[6];
    const float* att_b   = (const float*)d_in[7];
    const float* wW      = (const float*)d_in[8];
    const float* wb      = (const float*)d_in[9];
    const float* vW      = (const float*)d_in[10];
    const float* vb      = (const float*)d_in[11];
    const float* w_att_v = (const float*)d_in[12];
    const float* dec_Wih = (const float*)d_in[13];
    const float* dec_Whh = (const float*)d_in[14];
    const float* dec_b   = (const float*)d_in[15];
    const float* cls_W   = (const float*)d_in[16];
    const float* cls_b   = (const float*)d_in[17];
    float* logit = (float*)d_out;
    (void)in_sizes; (void)n_in; (void)out_size; (void)ws_size;

    char* ws = (char*)d_ws;
    size_t off = 0;
    auto take = [&](size_t bytes) -> char* {
        char* p = ws + off;
        off = (off + bytes + 255) & ~(size_t)255;
        return p;
    };
    float* att_h            = (float*)take((size_t)Bm * Tm * Dm * 4);        // 64 MB
    float* WihT             = (float*)take((size_t)K2 * G4 * 4);             // 2 MB
    float* WhhT             = (float*)take((size_t)Dm * G4 * 4);             // 1 MB
    float* vWT              = (float*)take((size_t)Dm * Dm * 4);
    float* dWihT            = (float*)take((size_t)K2 * G4 * 4);
    float* dWhhT            = (float*)take((size_t)Dm * G4 * 4);
    float* att_fea          = (float*)take((size_t)Bm * Sm * K2 * 4);        // 8 MB
    unsigned short* dout_bf = (unsigned short*)take((size_t)Bm * Sm * K3 * 2);
    unsigned short* clsW_bf = (unsigned short*)take((size_t)Vm * K3 * 2);
    unsigned short* eout_bf = (unsigned short*)take((size_t)Bm * Tm * Dm * 2); // 32 MB
    unsigned short* wW_bf   = (unsigned short*)take((size_t)Dm * Dm * 2);

    // ---- one-time prep: weight transposes (coalesced GEMV) + bf16 operand copies ----
    int n1 = G4 * K2, n2 = G4 * Dm, n3 = Dm * Dm;
    k_transpose<<<(n1 + 255) / 256, 256, 0, stream>>>(att_Wih, WihT, G4, K2);
    k_transpose<<<(n2 + 255) / 256, 256, 0, stream>>>(att_Whh, WhhT, G4, Dm);
    k_transpose<<<(n3 + 255) / 256, 256, 0, stream>>>(vW, vWT, Dm, Dm);
    k_transpose<<<(n1 + 255) / 256, 256, 0, stream>>>(dec_Wih, dWihT, G4, K2);
    k_transpose<<<(n2 + 255) / 256, 256, 0, stream>>>(dec_Whh, dWhhT, G4, Dm);
    int c1 = (Vm * K3) / 4, c2 = (Bm * Tm * Dm) / 4, c3 = (Dm * Dm) / 4;
    k_f32_to_bf16<<<(c1 + 255) / 256, 256, 0, stream>>>(cls_W, (unsigned int*)clsW_bf, c1);
    k_f32_to_bf16<<<(c2 + 255) / 256, 256, 0, stream>>>(eout, (unsigned int*)eout_bf, c2);
    k_f32_to_bf16<<<(c3 + 255) / 256, 256, 0, stream>>>(wW, (unsigned int*)wW_bf, c3);

    // ---- pipeline ----
    k_att_h<<<dim3((Bm * Tm) / 64), 256, 0, stream>>>(eout_bf, wW_bf, wb, att_h);
    k_att_scan<<<Bm, 256, 0, stream>>>(eout, x_mask, y, y_mask, emb,
                                       WihT, WhhT, att_b, vWT, vb, w_att_v,
                                       att_h, att_fea);
    k_dec_scan<<<Bm, 256, 0, stream>>>(att_fea, y_mask, dWihT, dWhhT, dec_b, dout_bf);
    k_cls<<<dim3((Bm * Sm) / 64, (Vm + 255) / 256), 256, 0, stream>>>(dout_bf, clsW_bf, cls_b, logit);
}
